// ANNEVO_51178830299718
// MI455X (gfx1250) — matrix operand order repaired and checked by reading, not compiled, not run on hardware
//
#include <hip/hip_runtime.h>
#include <hip/hip_bf16.h>
#include <stdint.h>
#include <math.h>

typedef _Float16 half_t;
typedef __attribute__((ext_vector_type(16))) _Float16 v16h;
typedef __attribute__((ext_vector_type(8)))  _Float16 v8h;
typedef __attribute__((ext_vector_type(8)))  float    v8f;

#define ACT_NONE 0
#define ACT_RELU 1
#define ACT_LRELU 2

// ---------------------------------------------------------------------------
// A-fragment loader (16x32 f16, ISA layout from CDNA5 ISA):
// lane m = lane%16, halves [0..7] = K kb..kb+7, [8..15] = K 16+kb..16+kb+7,
// kb = 0|8 by lane half. Fast path: two b128 loads + prefetch. Slow path:
// clamped addresses + v_cndmask zero-fill (no per-element branches).
// ---------------------------------------------------------------------------
__device__ __forceinline__
v16h frag_a(const half_t* __restrict__ A, int ldaM, int ldaK,
            int mrow, int k0, int kbA, int M, int K, bool fast)
{
    union { v16h v; v8h h8[2]; half_t h[16]; } u;
    if (fast && mrow < M && (k0 + 32) <= K) {
        const half_t* ap = A + (size_t)mrow * ldaM + k0;
        u.h8[0] = *(const v8h*)(ap + kbA);
        u.h8[1] = *(const v8h*)(ap + 16 + kbA);
        __builtin_prefetch(ap + 64, 0, 1);          // global_prefetch_b8
    } else {
        const int  mr  = mrow < M ? mrow : (M - 1);
        const bool mok = mrow < M;
#pragma unroll
        for (int j = 0; j < 16; ++j) {
            int kk = k0 + ((j < 8) ? (kbA + j) : (16 + kbA + (j - 8)));
            int kc = kk < K ? kk : (K - 1);
            half_t v = A[(size_t)mr * ldaM + (size_t)kc * ldaK];
            u.h[j] = (mok && kk < K) ? v : (half_t)0.0f;
        }
    }
    return u.v;
}

// ---------------------------------------------------------------------------
// Block-cooperative WMMA GEMM:  C[M,N] (+)= A[M,K] * B[K,N]  (f16 -> f32)
//   A element (m,k) at A + m*ldaM + k*ldaK
//   B element (k,n) at B + n*ldbN + k*ldbK
// 128 threads = 4 waves. Block tile 128x32. Per K-step the block stages the
// 32x32 B tile in LDS ([n][k], b128 stores) behind a split barrier; each wave
// computes a 32x32 tile: 2 A-frags x 2 B-frags -> 4 x v_wmma_f32_16x16x32_f16.
// Epilogue fuses accumulate / bias / f16-residual / (L)ReLU / f32+f16 stores.
// ---------------------------------------------------------------------------
__global__ __launch_bounds__(128)
void k_wmma_gemm(const half_t* __restrict__ A, int ldaM, int ldaK,
                 const half_t* __restrict__ Bw, int ldbN, int ldbK,
                 float* __restrict__ C32, int ldc32,
                 half_t* __restrict__ C16, int ldc16,
                 const float* __restrict__ bias,
                 const half_t* __restrict__ Res, int ldres,
                 int M, int N, int K, int accum, int act)
{
    __shared__ alignas(32) half_t Bs[32 * 32];      // [n][k] f16, 2 KB

    const int tid     = threadIdx.x;
    const int lane    = tid & 31;
    const int wave    = tid >> 5;
    const int tn      = blockIdx.x << 5;            // block N base (32)
    const int tm      = (blockIdx.y << 7) + (wave << 5); // this wave's 32-row strip
    const int half_hi = lane >> 4;
    const int l16     = lane & 15;
    const int kbA     = half_hi << 3;               // 0 | 8
    const int kbB     = half_hi << 4;               // 0 | 16
    const int mrow0   = tm + l16;
    const int mrow1   = mrow0 + 16;

    // staging decomposition: 1024 elements / 128 threads = 8 each
    const int sn  = (tid * 8) >> 5;                 // 0..31 (column)
    const int sk0 = (tid * 8) & 31;                 // 0,8,16,24 (k offset)

    auto loadC = [&](int rbase, int cbase) -> v8f {
        v8f a = {0.f, 0.f, 0.f, 0.f, 0.f, 0.f, 0.f, 0.f};
        if (accum) {
#pragma unroll
            for (int v = 0; v < 8; ++v) {
                int r  = rbase + v + (half_hi << 3);
                int c  = cbase + l16;
                int rc = r < M ? r : (M - 1);
                int cc = c < N ? c : (N - 1);
                float val = C32[(size_t)rc * ldc32 + cc];
                a[v] = (r < M && c < N) ? val : 0.f;
            }
        }
        return a;
    };
    v8f acc00 = loadC(tm, tn);
    v8f acc01 = loadC(tm, tn + 16);
    v8f acc10 = loadC(tm + 16, tn);
    v8f acc11 = loadC(tm + 16, tn + 16);

    const bool fastA = (ldaK == 1) && ((((uintptr_t)A)  & 15u) == 0) && ((ldaM & 7) == 0);
    const bool fastB = (ldbK == 1) && ((((uintptr_t)Bw) & 15u) == 0) && ((ldbN & 7) == 0);

    for (int k0 = 0; k0 < K; k0 += 32) {
        // ---- stage B tile -> LDS:  Bs[n*32+k] = B(k0+k, tn+n) ----
        {
            int gn = tn + sn;
            if (fastB && gn < N && (k0 + 32) <= K) {
                const half_t* bp = Bw + (size_t)gn * ldbN + k0 + sk0;
                *(v8h*)&Bs[sn * 32 + sk0] = *(const v8h*)bp;   // b128 ld/st
                __builtin_prefetch(bp + 64, 0, 1);
            } else {
                int gnc = gn < N ? gn : (N - 1);
#pragma unroll
                for (int j = 0; j < 8; ++j) {
                    int k  = sk0 + j;
                    int gk = k0 + k;
                    int gkc = gk < K ? gk : (K - 1);
                    half_t v = Bw[(size_t)gnc * ldbN + (size_t)gkc * ldbK];
                    Bs[sn * 32 + k] = (gn < N && gk < K) ? v : (half_t)0.0f;
                }
            }
        }
        __syncthreads();

        // ---- B fragments from LDS (ds_load_b128 pairs) ----
        union { v16h v; v8h h8[2]; } ub0, ub1;
        {
            const half_t* bp0 = &Bs[(size_t)l16 * 32 + kbA];
            const half_t* bp1 = &Bs[(size_t)(l16 + 16) * 32 + kbA];
            ub0.h8[0] = *(const v8h*)(bp0);
            ub0.h8[1] = *(const v8h*)(bp0 + 16);
            ub1.h8[0] = *(const v8h*)(bp1);
            ub1.h8[1] = *(const v8h*)(bp1 + 16);
        }

        // ---- A fragments from global ----
        v16h af0 = frag_a(A, ldaM, ldaK, mrow0, k0, kbA, M, K, fastA);
        v16h af1 = frag_a(A, ldaM, ldaK, mrow1, k0, kbA, M, K, fastA);

        acc00 = __builtin_amdgcn_wmma_f32_16x16x32_f16(false, af0, false, ub0.v,
                                                       (short)0, acc00, false, false);
        acc01 = __builtin_amdgcn_wmma_f32_16x16x32_f16(false, af0, false, ub1.v,
                                                       (short)0, acc01, false, false);
        acc10 = __builtin_amdgcn_wmma_f32_16x16x32_f16(false, af1, false, ub0.v,
                                                       (short)0, acc10, false, false);
        acc11 = __builtin_amdgcn_wmma_f32_16x16x32_f16(false, af1, false, ub1.v,
                                                       (short)0, acc11, false, false);
        __syncthreads();     // protect Bs before next stage
    }

    // ---- fused epilogue ----
    auto epi = [&](v8f acc, int rbase, int cbase) {
#pragma unroll
        for (int v = 0; v < 8; ++v) {
            int r = rbase + v + (half_hi << 3);
            int c = cbase + l16;
            if (r < M && c < N) {
                float x = acc[v];
                if (bias) x += bias[c];
                if (Res)  x += (float)Res[(size_t)r * ldres + c];
                if (act == ACT_RELU)       x = x > 0.f ? x : 0.f;
                else if (act == ACT_LRELU) x = x >= 0.f ? x : 0.1f * x;
                if (C32) C32[(size_t)r * ldc32 + c] = x;
                if (C16) C16[(size_t)r * ldc16 + c] = (half_t)x;
            }
        }
    };
    epi(acc00, tm, tn);
    epi(acc01, tm, tn + 16);
    epi(acc10, tm + 16, tn);
    epi(acc11, tm + 16, tn + 16);
}

// ---------------------------------------------------------------------------
// Elementwise / reduction helper kernels (wave32 VALU)
// ---------------------------------------------------------------------------
__global__ void k_f2h(const float* __restrict__ in, half_t* __restrict__ out, size_t n)
{
    size_t i  = (size_t)blockIdx.x * blockDim.x + threadIdx.x;
    size_t st = (size_t)gridDim.x * blockDim.x;
    for (; i < n; i += st) out[i] = (half_t)in[i];
}

// flat f32 [B*L*C] -> halo-padded f16 [B][h+L+h][C]
__global__ void k_f2h_halo(const float* __restrict__ in, half_t* __restrict__ out,
                           int Bn, int L, int C, int hOut)
{
    size_t n  = (size_t)Bn * L * C;
    size_t i  = (size_t)blockIdx.x * blockDim.x + threadIdx.x;
    size_t st = (size_t)gridDim.x * blockDim.x;
    for (; i < n; i += st) {
        int    c  = (int)(i % C);
        size_t rf = i / C;
        int    b  = (int)(rf / L);
        int    l  = (int)(rf % L);
        out[(((size_t)b * (L + 2 * hOut)) + hOut + l) * C + c] = (half_t)in[i];
    }
}

// inference BN (running stats 0/1) + LeakyReLU(0.1), halo->halo f16
__global__ void k_bn_act(const half_t* __restrict__ in, half_t* __restrict__ out,
                         const float* __restrict__ g, const float* __restrict__ bb,
                         int Bn, int L, int C, int hIn, int hOut)
{
    size_t n  = (size_t)Bn * L * C;
    size_t i  = (size_t)blockIdx.x * blockDim.x + threadIdx.x;
    size_t st = (size_t)gridDim.x * blockDim.x;
    const float rs = rsqrtf(1.f + 1e-5f);
    for (; i < n; i += st) {
        int    c  = (int)(i % C);
        size_t rf = i / C;
        int    b  = (int)(rf / L);
        int    l  = (int)(rf % L);
        float x = (float)in[(((size_t)b * (L + 2 * hIn)) + hIn + l) * C + c];
        x = x * (g[c] * rs) + bb[c];
        x = (x >= 0.f) ? x : 0.1f * x;
        out[(((size_t)b * (L + 2 * hOut)) + hOut + l) * C + c] = (half_t)x;
    }
}

// MaxPool1d(2) along positions, channel-last layout
__global__ void k_maxpool2(const half_t* __restrict__ in, half_t* __restrict__ out,
                           int Bn, int Lout, int C, int hIn, int hOut)
{
    size_t n  = (size_t)Bn * Lout * C;
    size_t i  = (size_t)blockIdx.x * blockDim.x + threadIdx.x;
    size_t st = (size_t)gridDim.x * blockDim.x;
    int Lin = 2 * Lout;
    for (; i < n; i += st) {
        int    c  = (int)(i % C);
        size_t rf = i / C;
        int    b  = (int)(rf / Lout);
        int    l  = (int)(rf % Lout);
        const half_t* ib = in + (((size_t)b * (Lin + 2 * hIn)) + hIn) * C;
        float a0 = (float)ib[(size_t)(2 * l) * C + c];
        float a1 = (float)ib[(size_t)(2 * l + 1) * C + c];
        out[(((size_t)b * (Lout + 2 * hOut)) + hOut + l) * C + c] =
            (half_t)(a0 > a1 ? a0 : a1);
    }
}

// sinusoidal positional encoding add (f16 in place)
__global__ void k_add_pe(half_t* __restrict__ h, int Bn, int T, int D)
{
    size_t n  = (size_t)Bn * T * D;
    size_t i  = (size_t)blockIdx.x * blockDim.x + threadIdx.x;
    size_t st = (size_t)gridDim.x * blockDim.x;
    for (; i < n; i += st) {
        int d = (int)(i % D);
        int t = (int)((i / D) % T);
        float freq = expf((float)(d & ~1) * (-9.210340371976184f / (float)D));
        float ang  = (float)t * freq;
        float pe   = (d & 1) ? cosf(ang) : sinf(ang);
        h[i] = (half_t)((float)h[i] + pe);
    }
}

// row softmax (with scale), f32 in -> f16 probs, one wave per row
__global__ __launch_bounds__(32)
void k_softmax_row(const float* __restrict__ S, half_t* __restrict__ P, int T, float scale)
{
    int row = blockIdx.x, lane = threadIdx.x;
    const float* sr = S + (size_t)row * T;
    float mx = -1e30f;
    for (int j = lane; j < T; j += 32) { float v = sr[j] * scale; mx = fmaxf(mx, v); }
    for (int o = 16; o; o >>= 1) mx = fmaxf(mx, __shfl_xor(mx, o, 32));
    float sum = 0.f;
    for (int j = lane; j < T; j += 32) sum += expf(sr[j] * scale - mx);
    for (int o = 16; o; o >>= 1) sum += __shfl_xor(sum, o, 32);
    float inv = 1.f / sum;
    half_t* pr = P + (size_t)row * T;
    for (int j = lane; j < T; j += 32) pr[j] = (half_t)(expf(sr[j] * scale - mx) * inv);
}

// LayerNorm over last dim, f32 in -> f16 out, one wave per row
__global__ __launch_bounds__(32)
void k_layernorm(const float* __restrict__ in, half_t* __restrict__ out,
                 const float* __restrict__ g, const float* __restrict__ b, int D)
{
    int row = blockIdx.x, lane = threadIdx.x;
    const float* x = in + (size_t)row * D;
    float s = 0.f;
    for (int j = lane; j < D; j += 32) s += x[j];
    for (int o = 16; o; o >>= 1) s += __shfl_xor(s, o, 32);
    float mean = s / (float)D;
    float v = 0.f;
    for (int j = lane; j < D; j += 32) { float d = x[j] - mean; v += d * d; }
    for (int o = 16; o; o >>= 1) v += __shfl_xor(v, o, 32);
    float inv = rsqrtf(v / (float)D + 1e-5f);
    half_t* orow = out + (size_t)row * D;
    for (int j = lane; j < D; j += 32)
        orow[j] = (half_t)((x[j] - mean) * inv * g[j] + b[j]);
}

// MoE gate: softmax(8) -> top2 -> combine weights + ti/tv outputs
__global__ void k_gate_topk(const float* __restrict__ logits, float* __restrict__ combine,
                            int* __restrict__ ti, float* __restrict__ tv, int tcount)
{
    int t = blockIdx.x * blockDim.x + threadIdx.x;
    if (t >= tcount) return;
    float l[8];
    float mx = -1e30f;
    for (int e = 0; e < 8; ++e) { l[e] = logits[(size_t)t * 8 + e]; mx = fmaxf(mx, l[e]); }
    float sum = 0.f;
    for (int e = 0; e < 8; ++e) { l[e] = expf(l[e] - mx); sum += l[e]; }
    float inv = 1.f / sum;
    for (int e = 0; e < 8; ++e) l[e] *= inv;
    int i0 = 0;
    for (int e = 1; e < 8; ++e) if (l[e] > l[i0]) i0 = e;
    int i1 = (i0 == 0) ? 1 : 0;
    for (int e = 0; e < 8; ++e) if (e != i0 && l[e] > l[i1]) i1 = e;
    ti[(size_t)t * 2 + 0] = i0; ti[(size_t)t * 2 + 1] = i1;
    tv[(size_t)t * 2 + 0] = l[i0]; tv[(size_t)t * 2 + 1] = l[i1];
    for (int e = 0; e < 8; ++e)
        combine[(size_t)t * 8 + e] = (e == i0 ? l[i0] : 0.f) + (e == i1 ? l[i1] : 0.f);
}

// acc[t,d] += combine[t,e] * eo[t,d]
__global__ void k_moe_accum(float* __restrict__ acc, const float* __restrict__ eo,
                            const float* __restrict__ comb, int e, size_t n, int D)
{
    size_t i  = (size_t)blockIdx.x * blockDim.x + threadIdx.x;
    size_t st = (size_t)gridDim.x * blockDim.x;
    for (; i < n; i += st) acc[i] += comb[(i / D) * 8 + e] * eo[i];
}

static inline dim3 gs1(size_t n, int bs) {
    size_t g = (n + (size_t)bs - 1) / bs; if (g < 1) g = 1; return dim3((unsigned)g);
}

// ---------------------------------------------------------------------------
extern "C" void kernel_launch(void* const* d_in, const int* in_sizes, int n_in,
                              void* d_out, int out_size, void* d_ws, size_t ws_size,
                              hipStream_t stream)
{
    (void)in_sizes; (void)n_in; (void)out_size; (void)ws_size;
    const int Bsz = 4, LINc = 10240, WINc = 8192, FLK = 1024, CHc = 64;
    const int Dm = 256, NHe = 8, Tt = 1280, DFFc = 1024, NEe = 8, NLl = 2, NCc = 8;

    int ii = 0;
    auto IN = [&]() -> const float* { return (const float*)d_in[ii++]; };
    const float* x_f     = IN();
    const float* conv1_w = IN(); const float* conv1_b = IN();
    const float* conv2_w = IN(); const float* conv2_b = IN();
    const float *cb_bng[3], *cb_bnb[3], *cb_c1w[3], *cb_c1b[3], *cb_c2w[3], *cb_c2b[3];
    for (int bi = 0; bi < 3; ++bi) {
        cb_bng[bi] = IN(); cb_bnb[bi] = IN(); cb_c1w[bi] = IN();
        cb_c1b[bi] = IN(); cb_c2w[bi] = IN(); cb_c2b[bi] = IN();
    }
    const float* qkv_w = IN(); const float* qkv_b = IN();
    const float* out_w = IN(); const float* out_b = IN();
    const float* ln1_g = IN(); const float* ln1_b = IN();
    const float* ff1_w = IN(); const float* ff1_b = IN();
    const float* ff2_w = IN(); const float* ff2_b = IN();
    const float* ln2_g = IN(); const float* ln2_b = IN();
    const float* gate_w = IN(); const float* gate_b = IN();
    const float* moe_w1 = IN(); const float* moe_b1 = IN();
    const float* moe_w2 = IN(); const float* moe_b2 = IN();
    const float *tb_tcw[3], *tb_tcb[3], *tb_bng[3], *tb_bnb[3], *tb_cw[3], *tb_cb[3];
    for (int bi = 0; bi < 3; ++bi) {
        tb_tcw[bi] = IN(); tb_tcb[bi] = IN(); tb_bng[bi] = IN();
        tb_bnb[bi] = IN(); tb_cw[bi]  = IN(); tb_cb[bi]  = IN();
    }
    const float* fc_w = IN(); const float* fc_b = IN();

    // bump allocator over workspace
    char* wsb = (char*)d_ws; size_t off = 0;
    auto ah = [&](size_t n) -> half_t* {
        off = (off + 255) & ~(size_t)255; half_t* p = (half_t*)(wsb + off);
        off += n * sizeof(half_t); return p; };
    auto af = [&](size_t n) -> float* {
        off = (off + 255) & ~(size_t)255; float* p = (float*)(wsb + off);
        off += n * sizeof(float); return p; };
    auto zero = [&](void* p, size_t bytes) { hipMemsetAsync(p, 0, bytes, stream); };

    auto gemm = [&](const half_t* A, int ldaM, int ldaK,
                    const half_t* Bw, int ldbN, int ldbK,
                    float* C32, int ldc32, half_t* C16, int ldc16,
                    const float* bias, const half_t* Res, int ldres,
                    int M, int N, int K, int accum, int act) {
        dim3 g((N + 31) / 32, (M + 127) / 128);
        k_wmma_gemm<<<g, dim3(128), 0, stream>>>(A, ldaM, ldaK, Bw, ldbN, ldbK,
                                                 C32, ldc32, C16, ldc16,
                                                 bias, Res, ldres, M, N, K, accum, act);
    };
    auto cvt = [&](const float* s, size_t n) -> half_t* {
        half_t* d = ah(n);
        k_f2h<<<gs1(n, 256), dim3(256), 0, stream>>>(s, d, n);
        return d; };

    // ---- weights -> f16 (persist) ----
    half_t* w_c1 = cvt(conv1_w, (size_t)64 * 4 * 9);
    half_t* w_c2 = cvt(conv2_w, (size_t)64 * 64);
    half_t *w_cb1[3], *w_cb2[3];
    for (int bi = 0; bi < 3; ++bi) {
        int cin = CHc * (bi + 1), cot = CHc * (bi + 2);
        w_cb1[bi] = cvt(cb_c1w[bi], (size_t)cot * cin * 3);
        w_cb2[bi] = cvt(cb_c2w[bi], (size_t)cot * cot);
    }
    half_t* w_qkv  = cvt(qkv_w,  (size_t)NLl * 3 * Dm * Dm);
    half_t* w_outp = cvt(out_w,  (size_t)NLl * Dm * Dm);
    half_t* w_ff1  = cvt(ff1_w,  (size_t)NLl * DFFc * Dm);
    half_t* w_ff2  = cvt(ff2_w,  (size_t)NLl * Dm * DFFc);
    half_t* w_gate = cvt(gate_w, (size_t)NEe * Dm);
    half_t* w_m1   = cvt(moe_w1, (size_t)NEe * DFFc * Dm);
    half_t* w_m2   = cvt(moe_w2, (size_t)NEe * Dm * DFFc);
    half_t *w_tc[3], *w_tbc[3];
    for (int bi = 0; bi < 3; ++bi) {
        int cin = CHc * (4 - bi), cot = CHc * (3 - bi);
        w_tc[bi]  = cvt(tb_tcw[bi], (size_t)cin * cot * 3);
        w_tbc[bi] = cvt(tb_cw[bi],  (size_t)cot * cot * 3);
    }
    half_t* w_fc = cvt(fc_w, (size_t)NCc * CHc);

    // persistent scratch
    float* CBf = af((size_t)Bsz * LINc * 128);                  // biggest f32 accum
    const size_t IOS = (size_t)Bsz * (LINc + 2) * CHc;          // max halo-padded IO
    half_t* IO0 = ah(IOS);
    half_t* IO1 = ah(IOS);

    // =========================== conv encoder ==============================
    half_t* xh = ah((size_t)Bsz * (LINc + 8) * 4);
    zero(xh, (size_t)Bsz * (LINc + 8) * 4 * sizeof(half_t));
    k_f2h_halo<<<gs1((size_t)Bsz * LINc * 4, 256), dim3(256), 0, stream>>>(
        x_f, xh, Bsz, LINc, 4, 4);

    half_t* c1h = ah((size_t)Bsz * LINc * CHc);
    // conv1: 4->64, k=9, pad=4 as 9 shifted tap-GEMMs per batch
    for (int b = 0; b < Bsz; ++b) {
        const half_t* xb = xh + (size_t)b * (LINc + 8) * 4;
        for (int t = 0; t < 9; ++t) {
            bool last = (t == 8);
            gemm(xb + t * 4, 4, 1, w_c1 + t, 36, 9,
                 CBf + (size_t)b * LINc * CHc, CHc,
                 last ? (c1h + (size_t)b * LINc * CHc) : nullptr, CHc,
                 last ? conv1_b : nullptr, nullptr, 0,
                 LINc, CHc, 4, (t > 0) ? 1 : 0, ACT_NONE);
        }
    }
    // conv2 (k=1) residual: t2 = t1 + conv2(t1)
    half_t* hcur = IO0;
    zero(hcur, IOS * sizeof(half_t));
    for (int b = 0; b < Bsz; ++b) {
        gemm(c1h + (size_t)b * LINc * CHc, CHc, 1, w_c2, CHc, 1,
             CBf + (size_t)b * LINc * CHc, CHc,
             hcur + ((size_t)b * (LINc + 2) + 1) * CHc, CHc,
             conv2_b, nullptr, 0, LINc, CHc, CHc, 1, ACT_NONE);
    }

    int Lc = LINc;
    size_t convMark = off;
    for (int bi = 0; bi < 3; ++bi) {
        off = convMark;
        int cin = CHc * (bi + 1), cot = CHc * (bi + 2);
        half_t* bnb = ah((size_t)Bsz * (Lc + 2) * cin);
        zero(bnb, (size_t)Bsz * (Lc + 2) * cin * sizeof(half_t));
        k_bn_act<<<gs1((size_t)Bsz * Lc * cin, 256), dim3(256), 0, stream>>>(
            hcur, bnb, cb_bng[bi], cb_bnb[bi], Bsz, Lc, cin, 1, 1);
        half_t* uh = ah((size_t)Bsz * Lc * cot);
        for (int b = 0; b < Bsz; ++b) {
            const half_t* Ab = bnb + ((size_t)b * (Lc + 2) + 1) * cin;
            for (int dk = 0; dk < 3; ++dk) {
                bool last = (dk == 2);
                gemm(Ab + (dk - 1) * cin, cin, 1, w_cb1[bi] + dk, cin * 3, 3,
                     CBf + (size_t)b * Lc * cot, cot,
                     last ? uh + (size_t)b * Lc * cot : nullptr, cot,
                     last ? cb_c1b[bi] : nullptr, nullptr, 0,
                     Lc, cot, cin, (dk > 0) ? 1 : 0, ACT_NONE);
            }
        }
        half_t* vh = ah((size_t)Bsz * Lc * cot);
        for (int b = 0; b < Bsz; ++b) {
            gemm(uh + (size_t)b * Lc * cot, cot, 1, w_cb2[bi], cot, 1,
                 CBf + (size_t)b * Lc * cot, cot,
                 vh + (size_t)b * Lc * cot, cot,
                 cb_c2b[bi], nullptr, 0, Lc, cot, cot, 1, ACT_NONE);
        }
        Lc >>= 1;
        half_t* nxt = (hcur == IO0) ? IO1 : IO0;
        int hOut = (bi < 2) ? 1 : 0;
        if (hOut) zero(nxt, IOS * sizeof(half_t));
        k_maxpool2<<<gs1((size_t)Bsz * Lc * cot, 256), dim3(256), 0, stream>>>(
            vh, nxt, Bsz, Lc, cot, 0, hOut);
        hcur = nxt;
    }
    off = convMark;

    // ============================ transformer ==============================
    k_add_pe<<<gs1((size_t)Bsz * Tt * Dm, 256), dim3(256), 0, stream>>>(hcur, Bsz, Tt, Dm);

    const int Mtok = Bsz * Tt;                       // 5120 tokens
    half_t* qkvh  = ah((size_t)Mtok * 3 * Dm);
    half_t* probs = ah((size_t)Tt * Tt);
    half_t* attno = ah((size_t)Mtok * Dm);
    half_t* ffh   = ah((size_t)Mtok * DFFc);
    float*  LN32  = af((size_t)Mtok * Dm);
    float*  SC32  = af((size_t)Tt * Tt);

    half_t* hh = hcur;
    const float scale = 0.17677669529663687f;        // 1/sqrt(dh=32)
    for (int l = 0; l < NLl; ++l) {
        gemm(hh, Dm, 1, w_qkv + (size_t)l * 3 * Dm * Dm, Dm, 1,
             nullptr, 0, qkvh, 3 * Dm, qkv_b + (size_t)l * 3 * Dm, nullptr, 0,
             Mtok, 3 * Dm, Dm, 0, ACT_NONE);
        for (int b = 0; b < Bsz; ++b) {
            for (int hd = 0; hd < NHe; ++hd) {
                const half_t* qb = qkvh + ((size_t)b * Tt) * (3 * Dm) + hd * 32;
                const half_t* kb = qb + Dm;
                const half_t* vb = qb + 2 * Dm;
                // S[t,s] = q[t,:] . k[s,:]  (K = dh = 32 -> one WMMA step)
                gemm(qb, 3 * Dm, 1, kb, 3 * Dm, 1, SC32, Tt, nullptr, 0,
                     nullptr, nullptr, 0, Tt, Tt, 32, 0, ACT_NONE);
                k_softmax_row<<<dim3(Tt), dim3(32), 0, stream>>>(SC32, probs, Tt, scale);
                // O[t,d] = sum_s P[t,s] V[s,d]  (B strided over s)
                gemm(probs, Tt, 1, vb, 1, 3 * Dm,
                     nullptr, 0, attno + ((size_t)b * Tt) * Dm + hd * 32, Dm,
                     nullptr, nullptr, 0, Tt, 32, Tt, 0, ACT_NONE);
            }
        }
        gemm(attno, Dm, 1, w_outp + (size_t)l * Dm * Dm, Dm, 1,
             LN32, Dm, nullptr, 0, out_b + (size_t)l * Dm, hh, Dm,
             Mtok, Dm, Dm, 0, ACT_NONE);
        half_t* h2 = (hh == IO0) ? IO1 : IO0;
        k_layernorm<<<dim3(Mtok), dim3(32), 0, stream>>>(
            LN32, h2, ln1_g + (size_t)l * Dm, ln1_b + (size_t)l * Dm, Dm);
        gemm(h2, Dm, 1, w_ff1 + (size_t)l * DFFc * Dm, Dm, 1,
             nullptr, 0, ffh, DFFc, ff1_b + (size_t)l * DFFc, nullptr, 0,
             Mtok, DFFc, Dm, 0, ACT_RELU);
        gemm(ffh, DFFc, 1, w_ff2 + (size_t)l * Dm * DFFc, DFFc, 1,
             LN32, Dm, nullptr, 0, ff2_b + (size_t)l * Dm, h2, Dm,
             Mtok, Dm, DFFc, 0, ACT_NONE);
        hh = (h2 == IO0) ? IO1 : IO0;
        k_layernorm<<<dim3(Mtok), dim3(32), 0, stream>>>(
            LN32, hh, ln2_g + (size_t)l * Dm, ln2_b + (size_t)l * Dm, Dm);
    }

    // ================================ MoE ==================================
    float* gate32 = af((size_t)Mtok * NEe);
    float* comb   = af((size_t)Mtok * NEe);
    float* eo32   = af((size_t)Mtok * Dm);
    float* macc   = af((size_t)Mtok * Dm);
    zero(macc, (size_t)Mtok * Dm * sizeof(float));
    gemm(hh, Dm, 1, w_gate, Dm, 1, gate32, NEe, nullptr, 0, gate_b, nullptr, 0,
         Mtok, NEe, Dm, 0, ACT_NONE);
    float* out_f  = (float*)d_out;
    int*   ti_out = (int*)(out_f + (size_t)Bsz * WINc * NCc);
    float* tv_out = out_f + (size_t)Bsz * WINc * NCc + (size_t)Mtok * 2;
    k_gate_topk<<<gs1((size_t)Mtok, 128), dim3(128), 0, stream>>>(
        gate32, comb, ti_out, tv_out, Mtok);
    for (int e = 0; e < NEe; ++e) {
        gemm(hh, Dm, 1, w_m1 + (size_t)e * DFFc * Dm, Dm, 1,
             nullptr, 0, ffh, DFFc, moe_b1 + (size_t)e * DFFc, nullptr, 0,
             Mtok, DFFc, Dm, 0, ACT_LRELU);
        gemm(ffh, DFFc, 1, w_m2 + (size_t)e * Dm * DFFc, DFFc, 1,
             eo32, Dm, nullptr, 0, moe_b2 + (size_t)e * Dm, nullptr, 0,
             Mtok, Dm, DFFc, 0, ACT_NONE);
        k_moe_accum<<<gs1((size_t)Mtok * Dm, 256), dim3(256), 0, stream>>>(
            macc, eo32, comb, e, (size_t)Mtok * Dm, Dm);
    }

    // ============================ task decoder =============================
    half_t* tcur = (hh == IO0) ? IO1 : IO0;
    zero(tcur, IOS * sizeof(half_t));
    k_f2h_halo<<<gs1((size_t)Mtok * Dm, 256), dim3(256), 0, stream>>>(
        macc, tcur, Bsz, Tt, Dm, 1);
    int Lt = Tt;
    size_t taskMark = off;
    for (int bi = 0; bi < 3; ++bi) {
        off = taskMark;
        int cin = CHc * (4 - bi);  // 256,192,128
        int cot = CHc * (3 - bi);  // 192,128,64
        int Lo = Lt * 2;
        half_t* tch = ah((size_t)Bsz * (Lo + 2) * cot);
        zero(tch, (size_t)Bsz * (Lo + 2) * cot * sizeof(half_t));
        for (int b = 0; b < Bsz; ++b) {
            const half_t* Ab  = tcur + ((size_t)b * (Lt + 2) + 1) * cin;
            half_t* C16b      = tch  + ((size_t)b * (Lo + 2) + 1) * cot;
            float*  C32b      = CBf  + (size_t)b * Lo * cot;
            // ConvTranspose1d(k=3,s=2,p=1,op=1): even out = x[m]*w[:,:,1]
            gemm(Ab, cin, 1, w_tc[bi] + 1, 3, cot * 3,
                 C32b, 2 * cot, C16b, 2 * cot, tb_tcb[bi], nullptr, 0,
                 Lt, cot, cin, 0, ACT_NONE);
            // odd out = x[m]*w[:,:,2] + x[m+1]*w[:,:,0]
            gemm(Ab, cin, 1, w_tc[bi] + 2, 3, cot * 3,
                 C32b + cot, 2 * cot, nullptr, 0, nullptr, nullptr, 0,
                 Lt, cot, cin, 0, ACT_NONE);
            gemm(Ab + cin, cin, 1, w_tc[bi] + 0, 3, cot * 3,
                 C32b + cot, 2 * cot, C16b + cot, 2 * cot, tb_tcb[bi], nullptr, 0,
                 Lt, cot, cin, 1, ACT_NONE);
        }
        half_t* bnT = ah((size_t)Bsz * (Lo + 2) * cot);
        zero(bnT, (size_t)Bsz * (Lo + 2) * cot * sizeof(half_t));
        k_bn_act<<<gs1((size_t)Bsz * Lo * cot, 256), dim3(256), 0, stream>>>(
            tch, bnT, tb_bng[bi], tb_bnb[bi], Bsz, Lo, cot, 1, 1);
        half_t* nxt = (tcur == IO0) ? IO1 : IO0;
        zero(nxt, IOS * sizeof(half_t));
        // residual conv k=3: h = h + conv(h)
        for (int b = 0; b < Bsz; ++b) {
            const half_t* Ab = bnT + ((size_t)b * (Lo + 2) + 1) * cot;
            half_t* Ob       = nxt + ((size_t)b * (Lo + 2) + 1) * cot;
            float*  C32b     = CBf + (size_t)b * Lo * cot;
            for (int dk = 0; dk < 3; ++dk) {
                bool last = (dk == 2);
                gemm(Ab + (dk - 1) * cot, cot, 1, w_tbc[bi] + dk, cot * 3, 3,
                     C32b, cot, last ? Ob : nullptr, cot,
                     last ? tb_cb[bi] : nullptr, last ? Ab : nullptr, cot,
                     Lo, cot, cot, (dk > 0) ? 1 : 0, ACT_NONE);
            }
        }
        tcur = nxt; Lt = Lo;
    }

    // final fc over the sliced window [FLANK, WIN+FLANK)
    for (int b = 0; b < Bsz; ++b) {
        const half_t* Ab = tcur + ((size_t)b * (LINc + 2) + 1) * CHc + (size_t)FLK * CHc;
        gemm(Ab, CHc, 1, w_fc, CHc, 1,
             out_f + (size_t)b * WINc * NCc, NCc, nullptr, 0,
             fc_b, nullptr, 0, WINc, NCc, CHc, 0, ACT_NONE);
    }
}
